// MDTA_FOR_VIDEO_Prompted_49237505081680
// MI455X (gfx1250) — matrix operand order repaired and checked by reading, not compiled, not run on hardware
//
#include <hip/hip_runtime.h>
#include <hip/hip_bf16.h>

// ---------------------------------------------------------------------------
// Types for CDNA5 WMMA
// ---------------------------------------------------------------------------
typedef __attribute__((ext_vector_type(16))) __bf16          v16bf;
typedef __attribute__((ext_vector_type(8)))  unsigned int    v8u;
typedef __attribute__((ext_vector_type(8)))  float           v8f;
typedef __attribute__((ext_vector_type(4)))  float           f4;

// pack two f32 -> packed bf16 pair (truncation) with one v_perm_b32
__device__ __forceinline__ unsigned int pkbf(float lo, float hi) {
    return __builtin_amdgcn_perm(__builtin_bit_cast(unsigned int, hi),
                                 __builtin_bit_cast(unsigned int, lo),
                                 0x07060302u);
}

#define BB 2          // batch
#define HH 128
#define WW 128
#define HWTOT (HH*WW)

// ---------------------------------------------------------------------------
// Implicit-GEMM conv via V_WMMA_F32_16X16X32_BF16 (wave32).
// Wave tile: 16 out-channels x 16 pixels along W. K = taps * Cin in 32-chunks.
// Layouts per CDNA5 ISA 7.12.2 (A: M=lane%16, K split by lane-half; B: N=lane%16,
// K=8*hi+e for e<8, 16+8*hi+(e-8) for e>=8; D: N=lane%16, M=r+8*hi).
// All indexing is 32-bit (uniform base + per-lane loop-invariant offset + constant*step)
// so loads lower to saddr/voffset + immediate forms; no 64-bit VALU addressing.
// A rows clamped (dup rows masked at store); B zero-padded via register selects
// only on edge tiles; K-tail zeroes B only.
// ---------------------------------------------------------------------------
template <int KS>
__global__ __launch_bounds__(128) void wmma_conv_kernel(
    const float* __restrict__ in, const float* __restrict__ wgt,
    const float* __restrict__ bias, const float* __restrict__ resid,
    float* __restrict__ out,
    int Cin, int Cout, int Hin, int Win, int Hout, int Wout,
    int pad, int inCtot, int inCoff, int outCtot, int outCoff,
    int residCtot, int relu)
{
    constexpr int K2 = KS * KS;
    const int lane = threadIdx.x & 31;
    const int wid  = threadIdx.x >> 5;
    const int lm   = lane & 15;
    const int hi   = lane >> 4;            // 0 or 1

    const int Mtiles = (Cout + 15) >> 4;
    const int Ntiles = (Wout + 15) >> 4;
    const int tilesPerImg = Mtiles * Ntiles * Hout;
    const int totalTiles  = tilesPerImg * BB;
    int tile = blockIdx.x * 4 + wid;       // wave-uniform guard
    if (tile >= totalTiles) return;

    int b  = tile / tilesPerImg;
    int t  = tile - b * tilesPerImg;
    int oy = t / (Mtiles * Ntiles);
    t -= oy * (Mtiles * Ntiles);
    int xt = t / Mtiles;
    int mt = t - xt * Mtiles;

    const int oc0   = mt * 16;
    const int x0    = xt * 16;
    const int plane = Hin * Win;
    int occ = oc0 + lm; if (occ > Cout - 1) occ = Cout - 1;   // clamped A row
    const int Cfull = Cin & ~31;

    const float* __restrict__ inImg =
        in + (size_t)(b * inCtot + inCoff) * plane;           // uniform base
    const unsigned wOffLane = (unsigned)occ * Cin * K2 + (unsigned)(hi * 8) * K2;

    v8f acc = {0.f,0.f,0.f,0.f,0.f,0.f,0.f,0.f};

    #pragma unroll
    for (int ty = 0; ty < KS; ++ty) {
        int iy = oy + ty - pad;
        if (iy < 0 || iy >= Hin) continue;          // wave-uniform skip
        #pragma unroll
        for (int tx = 0; tx < KS; ++tx) {
            int ixp = x0 + lm + tx - pad;           // B column pixel (N = lm)
            bool xok = (ixp >= 0) && (ixp < Win);
            int ixc = ixp < 0 ? 0 : (ixp > Win - 1 ? Win - 1 : ixp);
            // per-lane, loop-invariant 32-bit element offsets
            const unsigned aOffH = wOffLane + (unsigned)(ty * KS + tx);
            const unsigned bOffH = (unsigned)iy * Win + (unsigned)ixc
                                 + (unsigned)(hi * 8) * plane;
            // wave-uniform: whole 16-pixel span inside the row?
            const bool interior = (x0 + tx - pad >= 0) && (x0 + 15 + tx - pad < Win);

            int c0 = 0;
            if (interior) {
                for (; c0 < Cfull; c0 += 32) {
                    float fa[16], fb[16];
                    #pragma unroll
                    for (int e = 0; e < 16; ++e) {
                        int ke = (e < 8) ? e : e + 8;          // uniform constants
                        fa[e] = wgt[aOffH + (unsigned)((c0 + ke) * K2)];
                    }
                    #pragma unroll
                    for (int e = 0; e < 16; ++e)
                        fb[e] = inImg[bOffH + (unsigned)((c0 + ((e < 8) ? e : e + 8)) * plane)];
                    v8u apk, bpk;
                    #pragma unroll
                    for (int p = 0; p < 8; ++p) {
                        apk[p] = pkbf(fa[2 * p], fa[2 * p + 1]);
                        bpk[p] = pkbf(fb[2 * p], fb[2 * p + 1]);
                    }
                    acc = __builtin_amdgcn_wmma_f32_16x16x32_bf16(
                              false, __builtin_bit_cast(v16bf, apk),
                              false, __builtin_bit_cast(v16bf, bpk),
                              (short)0, acc, false, false);
                }
            } else {
                for (; c0 < Cfull; c0 += 32) {
                    float fa[16], fb[16];
                    #pragma unroll
                    for (int e = 0; e < 16; ++e) {
                        int ke = (e < 8) ? e : e + 8;
                        fa[e] = wgt[aOffH + (unsigned)((c0 + ke) * K2)];
                    }
                    #pragma unroll
                    for (int e = 0; e < 16; ++e) {
                        float v = inImg[bOffH + (unsigned)((c0 + ((e < 8) ? e : e + 8)) * plane)];
                        fb[e] = xok ? v : 0.f;                 // v_cndmask only
                    }
                    v8u apk, bpk;
                    #pragma unroll
                    for (int p = 0; p < 8; ++p) {
                        apk[p] = pkbf(fa[2 * p], fa[2 * p + 1]);
                        bpk[p] = pkbf(fb[2 * p], fb[2 * p + 1]);
                    }
                    acc = __builtin_amdgcn_wmma_f32_16x16x32_bf16(
                              false, __builtin_bit_cast(v16bf, apk),
                              false, __builtin_bit_cast(v16bf, bpk),
                              (short)0, acc, false, false);
                }
            }
            if (c0 < Cin) {                          // K tail: zero B beyond Cin
                const unsigned aOff = (unsigned)occ * Cin * K2 + (unsigned)(ty * KS + tx);
                const unsigned bOff = (unsigned)iy * Win + (unsigned)ixc;
                float fa[16], fb[16];
                #pragma unroll
                for (int e = 0; e < 16; ++e) {
                    int ke = ((e < 8) ? e : e + 8) + hi * 8;
                    int ci = c0 + ke; if (ci > Cin - 1) ci = Cin - 1;
                    fa[e] = wgt[aOff + (unsigned)(ci * K2)];
                    int cb = c0 + ke;                // true channel of elem e
                    int cc = cb > Cin - 1 ? Cin - 1 : cb;
                    float v = inImg[bOff + (unsigned)(cc * plane)];
                    fb[e] = (xok && cb < Cin) ? v : 0.f;
                }
                v8u apk, bpk;
                #pragma unroll
                for (int p = 0; p < 8; ++p) {
                    apk[p] = pkbf(fa[2 * p], fa[2 * p + 1]);
                    bpk[p] = pkbf(fb[2 * p], fb[2 * p + 1]);
                }
                acc = __builtin_amdgcn_wmma_f32_16x16x32_bf16(
                          false, __builtin_bit_cast(v16bf, apk),
                          false, __builtin_bit_cast(v16bf, bpk),
                          (short)0, acc, false, false);
            }
        }
    }

    // ---- store D: lane column ox, rows oc0 + r + hi*8
    int ox = x0 + lm;
    if (ox < Wout) {
        #pragma unroll
        for (int r = 0; r < 8; ++r) {
            int o = oc0 + r + hi * 8;
            if (o < Cout) {
                float v = acc[r];
                if (bias)  v += bias[o];
                if (resid) v += resid[(((size_t)(b * residCtot + o)) * Hout + oy) * Wout + ox];
                if (relu)  v = fmaxf(v, 0.f);
                out[(((size_t)(b * outCtot + outCoff + o)) * Hout + oy) * Wout + ox] = v;
            }
        }
    }
}

// ---------------------------------------------------------------------------
// LayerNorm over channel dim (C=32) of NCHW
// ---------------------------------------------------------------------------
__global__ __launch_bounds__(256) void layernorm_kernel(
    const float* __restrict__ x, const float* __restrict__ g,
    const float* __restrict__ bt, float* __restrict__ out)
{
    int i = blockIdx.x * 256 + threadIdx.x;
    if (i >= BB * HWTOT) return;
    int b = i / HWTOT, n = i % HWTOT;
    const float* xp = x + (size_t)b * 32 * HWTOT + n;
    float v[32], mu = 0.f;
    #pragma unroll
    for (int c = 0; c < 32; ++c) { v[c] = xp[(size_t)c * HWTOT]; mu += v[c]; }
    mu *= (1.f / 32.f);
    float var = 0.f;
    #pragma unroll
    for (int c = 0; c < 32; ++c) { float d = v[c] - mu; var += d * d; }
    var *= (1.f / 32.f);
    float inv = rsqrtf(var + 1e-5f);
    float* op = out + (size_t)b * 32 * HWTOT + n;
    #pragma unroll
    for (int c = 0; c < 32; ++c) op[(size_t)c * HWTOT] = (v[c] - mu) * inv * g[c] + bt[c];
}

// ---------------------------------------------------------------------------
// Grouped 3x3 conv (96ch, groups=32: 3 in -> 3 out), pad=1
// ---------------------------------------------------------------------------
__global__ __launch_bounds__(256) void groupconv3_kernel(
    const float* __restrict__ in, const float* __restrict__ wgt,
    float* __restrict__ out)
{
    int idx = blockIdx.x * 256 + threadIdx.x;
    int total = BB * 96 * HWTOT;
    if (idx >= total) return;
    int n = idx % HWTOT; int t = idx / HWTOT;
    int oc = t % 96; int b = t / 96;
    int y = n / WW, x = n % WW;
    int g = oc / 3, cinb = g * 3;
    float acc = 0.f;
    for (int c = 0; c < 3; ++c) {
        const float* pl = in + ((size_t)(b * 96 + cinb + c)) * HWTOT;
        const float* wp = wgt + ((size_t)(oc * 3 + c)) * 9;
        #pragma unroll
        for (int ty = 0; ty < 3; ++ty) {
            int iy = y + ty - 1; if (iy < 0 || iy >= HH) continue;
            #pragma unroll
            for (int tx = 0; tx < 3; ++tx) {
                int ix = x + tx - 1; if (ix < 0 || ix >= WW) continue;
                acc += pl[iy * WW + ix] * wp[ty * 3 + tx];
            }
        }
    }
    out[idx] = acc;
}

// ---------------------------------------------------------------------------
// Channel-attention Gram + norms per (b,head), reduced over n=HW:
// stats layout per bh: [32 G (d*4+e)][8 qq][4 kk]
// ---------------------------------------------------------------------------
__global__ __launch_bounds__(256) void gram_kernel(
    const float* __restrict__ q, const float* __restrict__ k,
    float* __restrict__ stats,
    int qCtot, int qCoff, int kCtot, int kCoff, int dq, int dk)
{
    int bh = blockIdx.x, b = bh / 8, h = bh % 8;
    int tid = threadIdx.x;
    float G[32], qq[8], kk[4];
    #pragma unroll
    for (int i = 0; i < 32; ++i) G[i] = 0.f;
    #pragma unroll
    for (int i = 0; i < 8; ++i) qq[i] = 0.f;
    #pragma unroll
    for (int i = 0; i < 4; ++i) kk[i] = 0.f;
    const float* qb = q + ((size_t)(b * qCtot + qCoff + h * dq)) * HWTOT;
    const float* kb = k + ((size_t)(b * kCtot + kCoff + h * dk)) * HWTOT;
    for (int n = tid; n < HWTOT; n += 256) {
        float qv[8], kv[4];
        #pragma unroll
        for (int d = 0; d < 8; ++d) qv[d] = (d < dq) ? qb[(size_t)d * HWTOT + n] : 0.f;
        #pragma unroll
        for (int e = 0; e < 4; ++e) kv[e] = (e < dk) ? kb[(size_t)e * HWTOT + n] : 0.f;
        #pragma unroll
        for (int d = 0; d < 8; ++d) {
            qq[d] += qv[d] * qv[d];
            #pragma unroll
            for (int e = 0; e < 4; ++e) G[d * 4 + e] += qv[d] * kv[e];
        }
        #pragma unroll
        for (int e = 0; e < 4; ++e) kk[e] += kv[e] * kv[e];
    }
    __shared__ float red[256];
    float* op = stats + (size_t)bh * 44;
    for (int i = 0; i < 44; ++i) {
        float v = (i < 32) ? G[i] : (i < 40 ? qq[i - 32] : kk[i - 40]);
        red[tid] = v; __syncthreads();
        for (int s = 128; s > 0; s >>= 1) {
            if (tid < s) red[tid] += red[tid + s];
            __syncthreads();
        }
        if (tid == 0) op[i] = red[0];
        __syncthreads();
    }
}

// attn[d,e] = softmax_e( G/(max(|q|,eps)max(|k|,eps)) * temp[h] )
__global__ void softmax_kernel(const float* __restrict__ stats,
                               const float* __restrict__ temp,
                               float* __restrict__ attnw, int dq, int dk)
{
    int bh = blockIdx.x, h = bh % 8;
    int d = threadIdx.x;
    if (d >= dq) return;
    const float* s = stats + (size_t)bh * 44;
    float qn = fmaxf(sqrtf(s[32 + d]), 1e-12f);
    float row[4], mx = -1e30f;
    for (int e = 0; e < dk; ++e) {
        float kn = fmaxf(sqrtf(s[40 + e]), 1e-12f);
        row[e] = s[d * 4 + e] / (qn * kn) * temp[h];
        mx = fmaxf(mx, row[e]);
    }
    float sum = 0.f;
    for (int e = 0; e < dk; ++e) { row[e] = expf(row[e] - mx); sum += row[e]; }
    for (int e = 0; e < dk; ++e) attnw[(size_t)bh * 32 + d * 4 + e] = row[e] / sum;
}

// out[b, h*dq+d, n] = sum_e attn[d,e] * v[b, h*dk+e, n]
__global__ __launch_bounds__(256) void attn_apply_kernel(
    const float* __restrict__ attnw, const float* __restrict__ v,
    float* __restrict__ out, int vCtot, int vCoff, int outCtot, int outCoff,
    int dq, int dk)
{
    const int segs = HWTOT / 256;
    int b = blockIdx.x / segs;
    int n = (blockIdx.x % segs) * 256 + threadIdx.x;
    __shared__ float sA[8 * 32];
    sA[threadIdx.x] = attnw[(size_t)(b * 8) * 32 + threadIdx.x];
    __syncthreads();
    for (int h = 0; h < 8; ++h) {
        float vv[4];
        for (int e = 0; e < 4; ++e)
            vv[e] = (e < dk) ? v[((size_t)(b * vCtot + vCoff + h * dk + e)) * HWTOT + n] : 0.f;
        for (int d = 0; d < dq; ++d) {
            float s = 0.f;
            for (int e = 0; e < dk; ++e) s += sA[h * 32 + d * 4 + e] * vv[e];
            out[((size_t)(b * outCtot + outCoff + h * dq + d)) * HWTOT + n] = s;
        }
    }
}

// AvgPool2d(2,2) on [B,64,H,W]
__global__ __launch_bounds__(256) void avgpool2_kernel(
    const float* __restrict__ in, float* __restrict__ out)
{
    const int Ho = HH / 2, Wo = WW / 2;
    int idx = blockIdx.x * 256 + threadIdx.x;
    int total = BB * 64 * Ho * Wo;
    if (idx >= total) return;
    int xo = idx % Wo; int t = idx / Wo;
    int yo = t % Ho; t /= Ho;
    int c = t % 64; int b = t / 64;
    const float* p = in + (((size_t)(b * 64 + c)) * HH + yo * 2) * WW + xo * 2;
    out[idx] = 0.25f * (p[0] + p[1] + p[WW] + p[WW + 1]);
}

// m *= sigmoid(nearest_resize(a, H, W))   (a is [B,166,62,62])
__global__ __launch_bounds__(256) void sigmul_kernel(
    float* __restrict__ m, const float* __restrict__ a)
{
    int idx = blockIdx.x * 256 + threadIdx.x;
    int total = BB * 166 * HWTOT;
    if (idx >= total) return;
    int x = idx % WW; int t = idx / WW;
    int y = t % HH; t /= HH;
    int c = t % 166; int b = t / 166;
    int ay = (y * 62) / HH, ax = (x * 62) / WW;
    float av = a[(((size_t)(b * 166 + c)) * 62 + ay) * 62 + ax];
    m[idx] *= 1.f / (1.f + expf(-av));
}

// Deformable conv (stride 1), Cin=32, Cout=32, groups=8 (4 in / 4 out per g)
__global__ __launch_bounds__(256) void deform_kernel(
    const float* __restrict__ x, const float* __restrict__ offs,
    const float* __restrict__ wgt, const float* __restrict__ bias,
    float* __restrict__ out, int k, int pad, int offC, int outCoff)
{
    int idx = blockIdx.x * 256 + threadIdx.x;
    int total = BB * 32 * HWTOT;
    if (idx >= total) return;
    int n = idx % HWTOT; int t = idx / HWTOT;
    int oc = t % 32; int b = t / 32;
    int y = n / WW, xx = n % WW;
    int g = oc >> 2, cinb = g * 4;
    int K2 = k * k;
    float acc = bias[oc];
    for (int j = 0; j < K2; ++j) {
        int ky = j / k, kx = j % k;
        float dy = offs[((size_t)(b * 166 + offC + 2 * j)) * HWTOT + n];
        float dx = offs[((size_t)(b * 166 + offC + 2 * j + 1)) * HWTOT + n];
        float py = (float)(y - pad + ky) + dy;
        float px = (float)(xx - pad + kx) + dx;
        float y0f = floorf(py), x0f = floorf(px);
        float wy = py - y0f, wx = px - x0f;
        int y0 = (int)y0f, x0 = (int)x0f;
        bool y0v = (y0 >= 0) && (y0 < HH), y1v = (y0 + 1 >= 0) && (y0 + 1 < HH);
        bool x0v = (x0 >= 0) && (x0 < WW), x1v = (x0 + 1 >= 0) && (x0 + 1 < WW);
        #pragma unroll
        for (int c = 0; c < 4; ++c) {
            const float* pl = x + ((size_t)(b * 32 + cinb + c)) * HWTOT;
            float v00 = (y0v && x0v) ? pl[y0 * WW + x0] : 0.f;
            float v01 = (y0v && x1v) ? pl[y0 * WW + x0 + 1] : 0.f;
            float v10 = (y1v && x0v) ? pl[(y0 + 1) * WW + x0] : 0.f;
            float v11 = (y1v && x1v) ? pl[(y0 + 1) * WW + x0 + 1] : 0.f;
            float s = v00 * (1.f - wy) * (1.f - wx) + v01 * (1.f - wy) * wx
                    + v10 * wy * (1.f - wx)         + v11 * wy * wx;
            acc += s * wgt[(((size_t)(oc * 4 + c)) * k + ky) * k + kx];
        }
    }
    out[((size_t)(b * 96 + outCoff + oc)) * HWTOT + n] = fmaxf(acc, 0.f);
}

// channel-slice copy for concatenation (vectorized b128)
__global__ __launch_bounds__(256) void copy_slice_kernel(
    const float* __restrict__ in, float* __restrict__ out,
    int C, int inCtot, int inCoff, int outCtot, int outCoff)
{
    const int HW4 = HWTOT / 4;
    int idx = blockIdx.x * 256 + threadIdx.x;
    int total = BB * C * HW4;
    if (idx >= total) return;
    int n = idx % HW4; int t = idx / HW4;
    int c = t % C; int b = t / C;
    const f4* ip = (const f4*)(in + ((size_t)(b * inCtot + inCoff + c)) * HWTOT);
    f4* op = (f4*)(out + ((size_t)(b * outCtot + outCoff + c)) * HWTOT);
    op[n] = ip[n];
}

// ---------------------------------------------------------------------------
extern "C" void kernel_launch(void* const* d_in, const int* in_sizes, int n_in,
                              void* d_out, int out_size, void* d_ws, size_t ws_size,
                              hipStream_t stream)
{
    const float* x      = (const float*)d_in[0];
    const float* y      = (const float*)d_in[1];
    const float* temp   = (const float*)d_in[2];
    const float* w_po   = (const float*)d_in[3];
    const float* w_lp1  = (const float*)d_in[4];
    const float* w_lp2  = (const float*)d_in[5];
    const float* ln_g   = (const float*)d_in[6];
    const float* ln_b   = (const float*)d_in[7];
    const float* temp_in= (const float*)d_in[8];
    const float* w_qkv  = (const float*)d_in[9];
    const float* w_qkvd = (const float*)d_in[10];
    const float* w_mproj= (const float*)d_in[11];
    const float* w_c3   = (const float*)d_in[12];
    const float* w_k2   = (const float*)d_in[13];
    const float* w_k3   = (const float*)d_in[14];
    const float* w_k4   = (const float*)d_in[15];
    const float* w_d3   = (const float*)d_in[16];
    const float* b_d3   = (const float*)d_in[17];
    const float* w_d5   = (const float*)d_in[18];
    const float* b_d5   = (const float*)d_in[19];
    const float* w_d7   = (const float*)d_in[20];
    const float* b_d7   = (const float*)d_in[21];
    const float* w_pw   = (const float*)d_in[22];
    const float* b_pw   = (const float*)d_in[23];

    const size_t S = (size_t)BB * 32 * HWTOT;   // 1,048,576 floats
    float* ws = (float*)d_ws;
    size_t cur = 0;
    auto alloc = [&](size_t n) { float* p = ws + cur; cur += n; return p; };
    float* xn     = alloc(S);                       // also reused as attn1 output
    float* cat96  = alloc(3 * S);                   // [l1 | l2 | xg]
    float* qkv    = alloc(3 * S);                   // reused as [f3|f5|f7]
    float* qkvd   = alloc(3 * S);                   // reused as attn2 output
    float* prompt = alloc(S);
    float* catxy  = alloc(2 * S);                   // reused as q_feat
    float* pooled = alloc((size_t)BB * 64 * 64 * 64);
    float* a62    = alloc((size_t)BB * 166 * 62 * 62);
    float* k3o    = alloc((size_t)BB * 166 * HWTOT);  // also holds k3*sigmoid(a)
    float* offs   = alloc((size_t)BB * 166 * HWTOT);
    float* kfeat  = alloc(S);
    float* stats  = alloc(1024);
    float* attnw  = alloc(1024);
    (void)ws_size; (void)in_sizes; (void)n_in; (void)out_size;

    auto conv = [&](const float* in, const float* w, const float* bias,
                    const float* resid, float* out,
                    int Cin, int Cout, int Hin, int Win, int Hout, int Wout,
                    int ks, int pad, int inCtot, int inCoff,
                    int outCtot, int outCoff, int residCtot, int relu) {
        int Mtiles = (Cout + 15) / 16, Ntiles = (Wout + 15) / 16;
        long total = (long)BB * Hout * Ntiles * Mtiles;
        int grid = (int)((total + 3) / 4);
        if (ks == 1)
            wmma_conv_kernel<1><<<grid, 128, 0, stream>>>(
                in, w, bias, resid, out, Cin, Cout, Hin, Win, Hout, Wout,
                pad, inCtot, inCoff, outCtot, outCoff, residCtot, relu);
        else
            wmma_conv_kernel<3><<<grid, 128, 0, stream>>>(
                in, w, bias, resid, out, Cin, Cout, Hin, Win, Hout, Wout,
                pad, inCtot, inCoff, outCtot, outCoff, residCtot, relu);
    };
    auto g1 = [](long n) { return (int)((n + 255) / 256); };

    // ---- Multi_Receptive_Attentive_Prompts -------------------------------
    layernorm_kernel<<<g1((long)BB * HWTOT), 256, 0, stream>>>(x, ln_g, ln_b, xn);
    conv(x, w_lp1, nullptr, nullptr, cat96, 32, 32, HH, WW, HH, WW, 1, 0, 32, 0, 96, 0, 0, 0);
    conv(x, w_lp2, nullptr, nullptr, cat96, 32, 32, HH, WW, HH, WW, 3, 1, 32, 0, 96, 32, 0, 0);
    conv(xn, w_qkv, nullptr, nullptr, qkv, 32, 96, HH, WW, HH, WW, 1, 0, 32, 0, 96, 0, 0, 0);
    groupconv3_kernel<<<g1((long)BB * 96 * HWTOT), 256, 0, stream>>>(qkv, w_qkvd, qkvd);
    gram_kernel<<<16, 256, 0, stream>>>(qkvd, qkvd, stats, 96, 0, 96, 32, 4, 4);
    softmax_kernel<<<16, 8, 0, stream>>>(stats, temp_in, attnw, 4, 4);
    attn_apply_kernel<<<BB * (HWTOT / 256), 256, 0, stream>>>(
        attnw, qkvd, xn /*attn1 out*/, 96, 64, 32, 0, 4, 4);
    conv(xn, w_mproj, nullptr, x, cat96, 32, 32, HH, WW, HH, WW, 1, 0, 32, 0, 96, 64, 32, 0);
    conv(cat96, w_c3, nullptr, nullptr, prompt, 96, 32, HH, WW, HH, WW, 3, 1, 96, 0, 32, 0, 0, 0);

    // ---- Temporal_Alignment: scconv --------------------------------------
    copy_slice_kernel<<<g1((long)BB * 32 * HWTOT / 4), 256, 0, stream>>>(x, catxy, 32, 32, 0, 64, 0);
    copy_slice_kernel<<<g1((long)BB * 32 * HWTOT / 4), 256, 0, stream>>>(y, catxy, 32, 32, 0, 64, 32);
    avgpool2_kernel<<<g1((long)BB * 64 * 64 * 64), 256, 0, stream>>>(catxy, pooled);
    conv(pooled, w_k2, nullptr, nullptr, a62, 64, 166, 64, 64, 62, 62, 3, 0, 64, 0, 166, 0, 0, 0);
    conv(catxy, w_k3, nullptr, nullptr, k3o, 64, 166, HH, WW, HH, WW, 3, 1, 64, 0, 166, 0, 0, 0);
    sigmul_kernel<<<g1((long)BB * 166 * HWTOT), 256, 0, stream>>>(k3o, a62);
    conv(k3o, w_k4, nullptr, nullptr, offs, 166, 166, HH, WW, HH, WW, 3, 1, 166, 0, 166, 0, 0, 0);

    // ---- deformable convs (relu fused), into 96-ch concat (reuse qkv) ----
    float* fcat = qkv;
    deform_kernel<<<g1((long)BB * 32 * HWTOT), 256, 0, stream>>>(x, offs, w_d3, b_d3, fcat, 3, 1, 0, 0);
    deform_kernel<<<g1((long)BB * 32 * HWTOT), 256, 0, stream>>>(x, offs, w_d5, b_d5, fcat, 5, 2, 18, 32);
    deform_kernel<<<g1((long)BB * 32 * HWTOT), 256, 0, stream>>>(x, offs, w_d7, b_d7, fcat, 7, 3, 68, 64);
    conv(fcat, w_pw, b_pw, nullptr, kfeat, 96, 32, HH, WW, HH, WW, 1, 0, 96, 0, 32, 0, 0, 0);

    // ---- prompted channel attention --------------------------------------
    float* qfeat = catxy;
    copy_slice_kernel<<<g1((long)BB * 32 * HWTOT / 4), 256, 0, stream>>>(x, qfeat, 32, 32, 0, 64, 0);
    copy_slice_kernel<<<g1((long)BB * 32 * HWTOT / 4), 256, 0, stream>>>(prompt, qfeat, 32, 32, 0, 64, 32);
    gram_kernel<<<16, 256, 0, stream>>>(qfeat, kfeat, stats, 64, 0, 32, 0, 8, 4);
    softmax_kernel<<<16, 8, 0, stream>>>(stats, temp, attnw, 8, 4);
    float* attn2o = qkvd;
    attn_apply_kernel<<<BB * (HWTOT / 256), 256, 0, stream>>>(
        attnw, x, attn2o, 32, 0, 64, 0, 8, 4);
    conv(attn2o, w_po, nullptr, nullptr, (float*)d_out, 64, 32, HH, WW, HH, WW,
         1, 0, 64, 0, 32, 0, 0, 0);
}
